// CrossAttentionHead_16664473109088
// MI455X (gfx1250) — compile-verified, matrix operand order checked by reading, not run on hardware
//
#include <hip/hip_runtime.h>
#include <hip/hip_bf16.h>

// ---------------------------------------------------------------------------
// CrossAttentionHead for MI455X (gfx1250, wave32, WMMA bf16 path)
//   Qt = Wq q^T + bq ; Kt = Wk k^T + bk ; Vt = Wv v^T + bv   (transposed outs)
//   num = Qt Kt^T / 64 ; attn = softmax(num) ; out = attn Vt^T
// bf16 GEMMs use double-buffered LDS fed by GLOBAL_LOAD_ASYNC_TO_LDS_B128.
// ---------------------------------------------------------------------------

typedef __bf16 bf16_t;
typedef bf16_t v16bf __attribute__((ext_vector_type(16)));
typedef float  v8f   __attribute__((ext_vector_type(8)));

#define S_DIM 4096
#define DIN   1024
#define DQ    1024
#define DK    4096

#define BT    64   // block tile (M and N)
#define KTS   32   // K step per WMMA
#define LDST  40   // LDS row stride in bf16 elements (80B, padded, 16B aligned)

__device__ __forceinline__ bf16_t f2bf(float f) { return (bf16_t)f; }

union FragU {
  v16bf v;
  uint4 q[2];
  bf16_t e[16];
};

// Async copy of 32 contiguous bytes (global -> LDS), two B128 transfers.
// LDS address comes from the VDST VGPR; the offset applies to both LDS and global address.
__device__ __forceinline__ void async_stage32(unsigned lds_off,
                                              unsigned long long base,
                                              unsigned voff) {
  asm volatile(
      "global_load_async_to_lds_b128 %0, %1, %2\n\t"
      "global_load_async_to_lds_b128 %0, %1, %2 offset:16"
      :
      : "v"(lds_off), "v"(voff), "s"(base)
      : "memory");
}
__device__ __forceinline__ void wait_async0() {
  asm volatile("s_wait_asynccnt 0x0" ::: "memory");
}

// Synchronous stage of a 64x32 tile (rows row0.., cols k0..k0+31) of a
// row-major [rows, ld] fp32 matrix into LDS as bf16 (register convert).
// 128 threads: thread t handles row t>>1, col half (t&1)*16.
__device__ __forceinline__ void stage_tile_f32(const void* __restrict__ g,
                                               int ld, int row0, int k0,
                                               bf16_t* __restrict__ lds,
                                               int tid) {
  const int row = tid >> 1;
  const int cb  = (tid & 1) << 4;  // 0 or 16
  FragU t;
  const float* p = (const float*)g + (size_t)(row0 + row) * ld + k0 + cb;
  __builtin_prefetch(p + 2 * KTS, 0, 0);  // next-next K tile -> global_prefetch
  const float4* p4 = (const float4*)p;
#pragma unroll
  for (int i = 0; i < 4; ++i) {
    float4 f = p4[i];
    t.e[i * 4 + 0] = f2bf(f.x);
    t.e[i * 4 + 1] = f2bf(f.y);
    t.e[i * 4 + 2] = f2bf(f.z);
    t.e[i * 4 + 3] = f2bf(f.w);
  }
  uint4* d = (uint4*)&lds[row * LDST + cb];
  d[0] = t.q[0];
  d[1] = t.q[1];
}

// D[m,n] = scale * sum_k A[m,k]*B[n,k]  (+ bias[m])
// A: [M,K] row-major (fp32 or bf16), B: [N,K] row-major (fp32 or bf16)
// Out: [M,N] row-major (bf16 or fp32). 128 threads = 4 waves, 64x64 tile,
// double-buffered LDS; bf16 inputs go through the async-to-LDS engine.
template <bool A_F32, bool B_F32, bool BIAS, bool OUT_BF16>
__global__ __launch_bounds__(128) void gemm_wmma_nt(
    const void* __restrict__ A, const void* __restrict__ B,
    const float* __restrict__ bias, void* __restrict__ Out, int M, int N,
    int K, float scale) {
  constexpr bool ASYNC = (!A_F32 && !B_F32);

  __shared__ bf16_t Alds[2][BT * LDST];
  __shared__ bf16_t Blds[2][BT * LDST];

  const int tid  = threadIdx.x;
  const int lane = tid & 31;
  const int wave = tid >> 5;
  const int bm = blockIdx.y * BT;
  const int bn = blockIdx.x * BT;
  const int wm = (wave & 1) * 32;
  const int wn = (wave >> 1) * 32;
  const int lr = lane & 15;
  const int kh = (lane >> 4) << 3;  // 0 or 8: K-half select per 16-bit layout

  // staging assignment: thread t -> row t>>1, column half (t&1)*16
  const int srow = tid >> 1;
  const int scb  = (tid & 1) << 4;

  auto stageA = [&](int buf, int k0) {
    if constexpr (ASYNC) {
      const unsigned voff =
          (unsigned)(((size_t)(bm + srow) * K + k0 + scb) * sizeof(bf16_t));
      async_stage32((unsigned)(size_t)&Alds[buf][srow * LDST + scb],
                    (unsigned long long)A, voff);
    } else {
      stage_tile_f32(A, K, bm, k0, Alds[buf], tid);
    }
  };
  auto stageB = [&](int buf, int k0) {
    if constexpr (ASYNC) {
      const unsigned voff =
          (unsigned)(((size_t)(bn + srow) * K + k0 + scb) * sizeof(bf16_t));
      async_stage32((unsigned)(size_t)&Blds[buf][srow * LDST + scb],
                    (unsigned long long)B, voff);
    } else {
      stage_tile_f32(B, K, bn, k0, Blds[buf], tid);
    }
  };

  v8f c[2][2] = {};

  stageA(0, 0);
  stageB(0, 0);
  if constexpr (ASYNC) wait_async0();
  __syncthreads();

  int p = 0;
  for (int k0 = 0; k0 < K; k0 += KTS) {
    if (k0 + KTS < K) {
      stageA(p ^ 1, k0 + KTS);
      stageB(p ^ 1, k0 + KTS);
    }

    FragU a0, a1, b0, b1;
    const bf16_t* ap0 = &Alds[p][(wm + lr) * LDST + kh];
    const bf16_t* ap1 = &Alds[p][(wm + 16 + lr) * LDST + kh];
    const bf16_t* bp0 = &Blds[p][(wn + lr) * LDST + kh];
    const bf16_t* bp1 = &Blds[p][(wn + 16 + lr) * LDST + kh];
    a0.q[0] = *(const uint4*)ap0; a0.q[1] = *(const uint4*)(ap0 + 16);
    a1.q[0] = *(const uint4*)ap1; a1.q[1] = *(const uint4*)(ap1 + 16);
    b0.q[0] = *(const uint4*)bp0; b0.q[1] = *(const uint4*)(bp0 + 16);
    b1.q[0] = *(const uint4*)bp1; b1.q[1] = *(const uint4*)(bp1 + 16);

    c[0][0] = __builtin_amdgcn_wmma_f32_16x16x32_bf16(false, a0.v, false, b0.v,
                                                      (short)0, c[0][0], false, false);
    c[0][1] = __builtin_amdgcn_wmma_f32_16x16x32_bf16(false, a0.v, false, b1.v,
                                                      (short)0, c[0][1], false, false);
    c[1][0] = __builtin_amdgcn_wmma_f32_16x16x32_bf16(false, a1.v, false, b0.v,
                                                      (short)0, c[1][0], false, false);
    c[1][1] = __builtin_amdgcn_wmma_f32_16x16x32_bf16(false, a1.v, false, b1.v,
                                                      (short)0, c[1][1], false, false);

    if constexpr (ASYNC) wait_async0();
    __syncthreads();
    p ^= 1;
  }

  // Epilogue. C/D layout: VGPR r, lanes 0-15 -> M=r, N=lane; lanes 16-31 -> M=8+r.
  const int nbase = bn + wn + (lane & 15);
  const int mtop  = ((lane >> 4) << 3);
#pragma unroll
  for (int ar = 0; ar < 2; ++ar) {
#pragma unroll
    for (int r = 0; r < 8; ++r) {
      const int m = bm + wm + ar * 16 + r + mtop;
      float bv = 0.0f;
      if constexpr (BIAS) bv = bias[m];
#pragma unroll
      for (int br = 0; br < 2; ++br) {
        const float val = c[ar][br][r] * scale + bv;
        const size_t idx = (size_t)m * N + nbase + br * 16;
        if constexpr (OUT_BF16)
          ((bf16_t*)Out)[idx] = f2bf(val);
        else
          ((float*)Out)[idx] = val;
      }
    }
  }
}

// Row softmax of x[rows, ncol] (fp32, read-only) -> bf16 attn.
__global__ __launch_bounds__(256) void softmax_rows_bf16(
    const float* __restrict__ x, bf16_t* __restrict__ attn, int ncol) {
  __shared__ float red[256];
  const int row = blockIdx.x;
  const int tid = threadIdx.x;
  const float* p = x + (size_t)row * ncol;

  float mx = -3.4e38f;
  for (int j = tid; j < ncol; j += 256) mx = fmaxf(mx, p[j]);
  red[tid] = mx;
  __syncthreads();
  for (int s = 128; s > 0; s >>= 1) {
    if (tid < s) red[tid] = fmaxf(red[tid], red[tid + s]);
    __syncthreads();
  }
  mx = red[0];
  __syncthreads();

  float sum = 0.0f;
  for (int j = tid; j < ncol; j += 256) sum += __expf(p[j] - mx);
  red[tid] = sum;
  __syncthreads();
  for (int s = 128; s > 0; s >>= 1) {
    if (tid < s) red[tid] += red[tid + s];
    __syncthreads();
  }
  const float inv = 1.0f / red[0];

  bf16_t* arow = attn + (size_t)row * ncol;
  for (int j = tid; j < ncol; j += 256) arow[j] = f2bf(__expf(p[j] - mx) * inv);
}

extern "C" void kernel_launch(void* const* d_in, const int* in_sizes, int n_in,
                              void* d_out, int out_size, void* d_ws,
                              size_t ws_size, hipStream_t stream) {
  (void)in_sizes; (void)n_in; (void)out_size; (void)ws_size;
  const float* q  = (const float*)d_in[0];
  const float* k  = (const float*)d_in[1];
  const float* v  = (const float*)d_in[2];
  const float* Wq = (const float*)d_in[3];
  const float* bq = (const float*)d_in[4];
  const float* Wk = (const float*)d_in[5];
  const float* bk = (const float*)d_in[6];
  const float* Wv = (const float*)d_in[7];
  const float* bv = (const float*)d_in[8];
  float* out = (float*)d_out;

  char* ws = (char*)d_ws;
  bf16_t* Qt    = (bf16_t*)(ws);                             // [DQ, S]   8 MB
  bf16_t* Kt    = (bf16_t*)(ws + (size_t)(8u)  * 1048576u);  // [DK, S]  32 MB
  bf16_t* Vt    = (bf16_t*)(ws + (size_t)(40u) * 1048576u);  // [DK, S]  32 MB
  bf16_t* attnb = (bf16_t*)(ws + (size_t)(72u) * 1048576u);  // [DQ, DK]  8 MB

  const dim3 blk(128);

  // Projections (transposed outputs): Pt = W · X^T + b
  gemm_wmma_nt<true, true, true, true>
      <<<dim3(S_DIM / BT, DQ / BT), blk, 0, stream>>>(Wq, q, bq, Qt, DQ, S_DIM,
                                                      DIN, 1.0f);
  gemm_wmma_nt<true, true, true, true>
      <<<dim3(S_DIM / BT, DK / BT), blk, 0, stream>>>(Wk, k, bk, Kt, DK, S_DIM,
                                                      DIN, 1.0f);
  gemm_wmma_nt<true, true, true, true>
      <<<dim3(S_DIM / BT, DK / BT), blk, 0, stream>>>(Wv, v, bv, Vt, DK, S_DIM,
                                                      DIN, 1.0f);

  // num = (Qt · Kt^T) / sqrt(DK), fp32 scratch in d_out
  gemm_wmma_nt<false, false, false, false>
      <<<dim3(DK / BT, DQ / BT), blk, 0, stream>>>(Qt, Kt, nullptr, out, DQ,
                                                   DK, S_DIM, 1.0f / 64.0f);

  // attn = softmax(num) rows -> bf16
  softmax_rows_bf16<<<dim3(DQ), dim3(256), 0, stream>>>(out, attnb, DK);

  // out = attn · Vt^T
  gemm_wmma_nt<false, false, false, false>
      <<<dim3(DK / BT, DQ / BT), blk, 0, stream>>>(attnb, Vt, nullptr, out, DQ,
                                                   DK, S_DIM, 1.0f);
}
